// MultiHeadAttention_10780367912960
// MI455X (gfx1250) — matrix operand order repaired and checked by reading, not compiled, not run on hardware
//
#include <hip/hip_runtime.h>

// ---------------------------------------------------------------------------
// MHA forward for B=4, T=2048, C=1024, H=16 (d=64) on gfx1250 (CDNA5, wave32)
// All matmuls via v_wmma_f32_16x16x32_f16 (f16 operands, f32 accumulate).
// Attention computed transposed (S^T = K Q^T, O^T = V^T P^T) so softmax
// stats are per-lane scalars (queries live on the lane axis of C/D tiles).
// ---------------------------------------------------------------------------

typedef __attribute__((ext_vector_type(16))) _Float16 v16h;
typedef __attribute__((ext_vector_type(8)))  float    v8f;

constexpr int kB  = 4;
constexpr int kT  = 2048;
constexpr int kC  = 1024;
constexpr int kH  = 16;
constexpr int kD  = 64;          // head dim
constexpr int kN3 = 3 * kC;      // 3072
constexpr int kM  = kB * kT;     // 8192 rows
constexpr int LSTR = 72;         // padded LDS stride (halves) for 64-wide rows

union FragU { v16h h; unsigned int u[8]; };

// A-fragment (16x32 f16). lane L -> row m = L&15, half-group hh = L>>4.
// VGPR v holds f16 pair at K = 16*hh + 2v.
__device__ inline v16h frag_a_lds(const _Float16* base, int stride, int lane, int koff) {
  int m = lane & 15, hh = lane >> 4;
  FragU f;
#pragma unroll
  for (int v = 0; v < 8; ++v) {
    int k = 16 * hh + 2 * v;
    f.u[v] = *(const unsigned int*)(base + m * stride + koff + k);
  }
  return f.h;
}

// B-fragment (32x16 f16) from an N-major tile Bt[n][k]. lane L -> col n = L&15,
// K-group g = L>>4. VGPR v holds pair at K = 16*g + 2v.
__device__ inline v16h frag_b_lds(const _Float16* base, int stride, int lane, int koff) {
  int n = lane & 15, g = lane >> 4;
  FragU f;
#pragma unroll
  for (int v = 0; v < 8; ++v) {
    int k = 16 * g + 2 * v;
    f.u[v] = *(const unsigned int*)(base + n * stride + koff + k);
  }
  return f.h;
}

// ---------------------------------------------------------------------------
// WMMA GEMM: out[M,N] = A[M,K] @ Bt[N,K]^T + bias[N]
// Block tile 128x128, 8 waves (4 x 2), wave tile 32x64, K-step 64.
// ---------------------------------------------------------------------------
template <int OUTF32>
__global__ __launch_bounds__(256) void wmma_gemm(
    const _Float16* __restrict__ A,    // [M,K] row major
    const _Float16* __restrict__ Bt,   // [N,K] row major (transposed weights)
    const float*    __restrict__ bias, // [N]
    _Float16* __restrict__ outH,
    float*    __restrict__ outF,
    int M, int N, int K)
{
  __shared__ __align__(16) _Float16 sA[128 * LSTR];
  __shared__ __align__(16) _Float16 sB[128 * LSTR];

  const int tid  = threadIdx.x;
  const int lane = tid & 31, wid = tid >> 5;
  const int wm = wid & 3, wn = wid >> 2;      // wave tile origin: (32*wm, 64*wn)
  const int m0 = blockIdx.y * 128, n0 = blockIdx.x * 128;

  v8f c[2][4];
#pragma unroll
  for (int i = 0; i < 2; ++i)
#pragma unroll
    for (int j = 0; j < 4; ++j) c[i][j] = v8f{};

  for (int kb = 0; kb < K; kb += 64) {
    // A and B tiles: 128 rows x 64 halves each = 1024 uint4 each; 4 per thread
#pragma unroll
    for (int i = 0; i < 4; ++i) {
      int idx = i * 256 + tid;
      int row = idx >> 3, c8 = idx & 7;
      *(uint4*)(sA + row * LSTR + c8 * 8) =
          *(const uint4*)(A + (size_t)(m0 + row) * K + kb + c8 * 8);
      *(uint4*)(sB + row * LSTR + c8 * 8) =
          *(const uint4*)(Bt + (size_t)(n0 + row) * K + kb + c8 * 8);
    }
    __syncthreads();

#pragma unroll
    for (int kc = 0; kc < 2; ++kc) {
      v16h a0 = frag_a_lds(sA + (wm * 32) * LSTR,      LSTR, lane, kc * 32);
      v16h a1 = frag_a_lds(sA + (wm * 32 + 16) * LSTR, LSTR, lane, kc * 32);
#pragma unroll
      for (int j = 0; j < 4; ++j) {
        v16h bj = frag_b_lds(sB + (wn * 64 + j * 16) * LSTR, LSTR, lane, kc * 32);
        c[0][j] = __builtin_amdgcn_wmma_f32_16x16x32_f16(false, a0, false, bj, (short)0, c[0][j], false, false);
        c[1][j] = __builtin_amdgcn_wmma_f32_16x16x32_f16(false, a1, false, bj, (short)0, c[1][j], false, false);
      }
    }
    __syncthreads();
  }

  // C/D layout: VGPR r, lane L -> row r + 8*(L>>4), col L&15
  const int nl = lane & 15, g = lane >> 4;
#pragma unroll
  for (int j = 0; j < 4; ++j) {
    int col = n0 + wn * 64 + j * 16 + nl;
    float bv = bias[col];
#pragma unroll
    for (int i = 0; i < 2; ++i) {
      int rowb = m0 + wm * 32 + i * 16 + 8 * g;
#pragma unroll
      for (int r = 0; r < 8; ++r) {
        float val = c[i][j][r] + bv;
        if (OUTF32) outF[(size_t)(rowb + r) * N + col] = val;
        else        outH[(size_t)(rowb + r) * N + col] = (_Float16)val;
      }
    }
  }
}

// ---------------------------------------------------------------------------
// Flash attention (transposed): block = 64 query rows of one (b,h), 4 waves,
// each wave owns 16 query rows; 64-key tiles staged in LDS per iteration.
//   S^T = K(64x64kd) @ Q^T          -> queries on lane axis
//   O^T = V^T(64d x 64key) @ P^T    -> alpha rescale is per-lane scalar
// ---------------------------------------------------------------------------
__global__ __launch_bounds__(128) void flash_attn(
    const _Float16* __restrict__ qkvh,  // [B*T, 3C]: q | k | v per row
    _Float16* __restrict__ attnh,       // [B*T, C] (col = h*64 + d)
    const int* __restrict__ decoderp)
{
  __shared__ __align__(16) _Float16 sK [64 * LSTR];     // [key][kd]
  __shared__ __align__(16) _Float16 sVt[64 * LSTR];     // [d][key]
  __shared__ __align__(16) _Float16 sP [4][16 * LSTR];  // per-wave P^T as [q][key]

  const int tid = threadIdx.x, lane = tid & 31, w = tid >> 5;
  const int q0 = blockIdx.x * 64;
  const int h  = blockIdx.y, b = blockIdx.z;
  const int qw = q0 + w * 16;
  const int nl = lane & 15, g = lane >> 4;
  const int dec  = decoderp[0];
  const int trow = qw + nl;            // this lane's query row

  // Q held as B-fragments (natural [q][kd] layout, pairs contiguous)
  const _Float16* qbase = qkvh + (size_t)(b * kT + qw) * kN3 + h * kD;
  v16h qb0, qb1;
  {
    FragU f0, f1;
#pragma unroll
    for (int v = 0; v < 8; ++v) {
      int k = 16 * g + 2 * v;
      f0.u[v] = *(const unsigned int*)(qbase + (size_t)nl * kN3 + k);
      f1.u[v] = *(const unsigned int*)(qbase + (size_t)nl * kN3 + 32 + k);
    }
    qb0 = f0.h; qb1 = f1.h;
  }

  v8f o[4];
#pragma unroll
  for (int dc = 0; dc < 4; ++dc) o[dc] = v8f{};
  float mi = -3.0e38f, li = 0.0f;
  const float scale = 32.0f;           // reference quirk: qk * sqrt(C), C=1024

  const int kend = dec ? (q0 + 64) : kT;
  for (int k0 = 0; k0 < kend; k0 += 64) {
    // K tile [64][64]: 512 uint4, 4 per thread
#pragma unroll
    for (int i = 0; i < 4; ++i) {
      int idx = i * 128 + tid;
      int row = idx >> 3, c8 = idx & 7;
      *(uint4*)(sK + row * LSTR + c8 * 8) =
          *(const uint4*)(qkvh + (size_t)(b * kT + k0 + row) * kN3 + kC + h * kD + c8 * 8);
    }
    // V tile transposed into [d][key]; vectorized global read, b16 scatter
    {
      int vr = tid >> 1, vh = tid & 1;
      const _Float16* vs = qkvh + (size_t)(b * kT + k0 + vr) * kN3 + 2 * kC + h * kD + vh * 32;
      union { uint4 q[4]; _Float16 e[32]; } u;
      u.q[0] = ((const uint4*)vs)[0];
      u.q[1] = ((const uint4*)vs)[1];
      u.q[2] = ((const uint4*)vs)[2];
      u.q[3] = ((const uint4*)vs)[3];
#pragma unroll
      for (int e = 0; e < 32; ++e)
        sVt[(vh * 32 + e) * LSTR + vr] = u.e[e];
    }
    __syncthreads();

    // S^T tiles: 4 key-groups of 16; K-dim 64 = 2 chained WMMAs each
    v8f st[4];
#pragma unroll
    for (int kg = 0; kg < 4; ++kg) {
      v8f s{};
      s = __builtin_amdgcn_wmma_f32_16x16x32_f16(false, frag_a_lds(sK + kg * 16 * LSTR, LSTR, lane, 0),  false, qb0, (short)0, s, false, false);
      s = __builtin_amdgcn_wmma_f32_16x16x32_f16(false, frag_a_lds(sK + kg * 16 * LSTR, LSTR, lane, 32), false, qb1, (short)0, s, false, false);
      st[kg] = s;
    }

    // Online softmax: this lane owns query column `trow`; its 32 key scores
    // (keys k0 + kg*16 + r + 8g) are local registers. One shfl to merge halves.
    float ev[4][8];
    float mx = -3.0e38f;
#pragma unroll
    for (int kg = 0; kg < 4; ++kg)
#pragma unroll
      for (int r = 0; r < 8; ++r) {
        int key = k0 + kg * 16 + r + 8 * g;
        float e = st[kg][r] * scale;
        if (dec && key > trow) e = -3.0e38f;
        ev[kg][r] = e;
        mx = fmaxf(mx, e);
      }
    mx = fmaxf(mx, __shfl_xor(mx, 16, 32));
    float mnew  = fmaxf(mi, mx);
    float alpha = __expf(mi - mnew);
    mi = mnew;

    float rs = 0.0f;
#pragma unroll
    for (int kg = 0; kg < 4; ++kg)
#pragma unroll
      for (int r = 0; r < 8; ++r) {
        float p = __expf(ev[kg][r] - mnew);
        rs += p;
        sP[w][nl * LSTR + kg * 16 + r + 8 * g] = (_Float16)p;  // P^T as [q][key]
      }
    rs += __shfl_xor(rs, 16, 32);
    li = li * alpha + rs;

#pragma unroll
    for (int dc = 0; dc < 4; ++dc)
#pragma unroll
      for (int r = 0; r < 8; ++r) o[dc][r] *= alpha;

    // O^T += V^T @ P^T : A = V^T chunks (16 d-rows), B = P^T (same-wave LDS,
    // in-order DS guarantees visibility)
    v16h pb0 = frag_b_lds(sP[w], LSTR, lane, 0);
    v16h pb1 = frag_b_lds(sP[w], LSTR, lane, 32);
#pragma unroll
    for (int dc = 0; dc < 4; ++dc) {
      o[dc] = __builtin_amdgcn_wmma_f32_16x16x32_f16(false, frag_a_lds(sVt + dc * 16 * LSTR, LSTR, lane, 0),  false, pb0, (short)0, o[dc], false, false);
      o[dc] = __builtin_amdgcn_wmma_f32_16x16x32_f16(false, frag_a_lds(sVt + dc * 16 * LSTR, LSTR, lane, 32), false, pb1, (short)0, o[dc], false, false);
    }
    __syncthreads();
  }

  // O^T layout: row = d = dc*16 + r + 8g, col = q = nl. Store row `trow`.
  float inv = 1.0f / li;
  _Float16* dst = attnh + (size_t)(b * kT + trow) * kC + h * kD;
#pragma unroll
  for (int dc = 0; dc < 4; ++dc)
#pragma unroll
    for (int r = 0; r < 8; ++r)
      dst[dc * 16 + 8 * g + r] = (_Float16)(o[dc][r] * inv);
}

// ---------------------------------------------------------------------------
// Conversion helpers
// ---------------------------------------------------------------------------
__global__ void cvt_f32_f16(const float* __restrict__ s, _Float16* __restrict__ d, int n) {
  int i = blockIdx.x * 256 + threadIdx.x;
  if (i < n) d[i] = (_Float16)s[i];
}

// s[R][Cc] (f32) -> d[Cc][R] (f16)
__global__ void transpose_f32_f16(const float* __restrict__ s, _Float16* __restrict__ d,
                                  int R, int Cc) {
  long i = (long)blockIdx.x * 256 + threadIdx.x;
  if (i < (long)R * Cc) {
    int r = (int)(i / Cc), c = (int)(i % Cc);
    d[(size_t)c * R + r] = (_Float16)s[i];
  }
}

// ---------------------------------------------------------------------------
extern "C" void kernel_launch(void* const* d_in, const int* in_sizes, int n_in,
                              void* d_out, int out_size, void* d_ws, size_t ws_size,
                              hipStream_t stream)
{
  (void)in_sizes; (void)n_in; (void)out_size; (void)ws_size;
  const float* x    = (const float*)d_in[0];
  const float* Wqkv = (const float*)d_in[1];
  const float* bqkv = (const float*)d_in[2];
  const float* Wo   = (const float*)d_in[3];
  const float* bo   = (const float*)d_in[4];
  const int*   dec  = (const int*)d_in[5];
  float* out = (float*)d_out;

  char* ws = (char*)d_ws;
  size_t off = 0;
  auto walloc = [&](size_t bytes) {
    void* p = ws + off;
    off += (bytes + 255) & ~(size_t)255;
    return p;
  };
  _Float16* xh    = (_Float16*)walloc((size_t)kM * kC * 2);     // 16 MB
  _Float16* WqkvT = (_Float16*)walloc((size_t)kN3 * kC * 2);    //  6 MB
  _Float16* WoT   = (_Float16*)walloc((size_t)kC * kC * 2);     //  2 MB
  _Float16* qkvh  = (_Float16*)walloc((size_t)kM * kN3 * 2);    // 48 MB
  _Float16* attnh = (_Float16*)walloc((size_t)kM * kC * 2);     // 16 MB

  cvt_f32_f16<<<(kM * kC + 255) / 256, 256, 0, stream>>>(x, xh, kM * kC);
  transpose_f32_f16<<<(kC * kN3 + 255) / 256, 256, 0, stream>>>(Wqkv, WqkvT, kC, kN3);
  transpose_f32_f16<<<(kC * kC + 255) / 256, 256, 0, stream>>>(Wo, WoT, kC, kC);

  dim3 g1(kN3 / 128, kM / 128);
  wmma_gemm<0><<<g1, 256, 0, stream>>>(xh, WqkvT, bqkv, qkvh, nullptr, kM, kN3, kC);

  dim3 g2(kT / 64, kH, kB);
  flash_attn<<<g2, 128, 0, stream>>>(qkvh, attnh, dec);

  dim3 g3(kC / 128, kM / 128);
  wmma_gemm<1><<<g3, 256, 0, stream>>>(attnh, WoT, bo, nullptr, out, kM, kC, kC);
}
